// MultiheadAttention_22892175688164
// MI455X (gfx1250) — matrix operand order repaired and checked by reading, not compiled, not run on hardware
//
#include <hip/hip_runtime.h>
#include <hip/hip_bf16.h>

// MHA for MI455X (gfx1250): bf16 WMMA (v_wmma_f32_16x16x32_bf16), wave32,
// 8 waves / 256-thread blocks, TDM (tensor_load_to_lds) tile staging with
// double-buffered LDS, register softmax.

typedef __bf16 bf16_t;
typedef __attribute__((ext_vector_type(16))) __bf16 v16bf;
typedef __attribute__((ext_vector_type(8)))  float  v8f;
typedef __attribute__((ext_vector_type(4)))  unsigned int u32x4;
typedef __attribute__((ext_vector_type(4)))  int i32x4;
typedef __attribute__((ext_vector_type(8)))  int i32x8;

#define D_MODEL   1024
#define KEY_SIZE  1024
#define NUM_HEADS 16
#define BATCH     8
#define SEQ       4096
#define MROWS     (BATCH * SEQ)      /* 32768 */
#define SQ        256                /* S / NUM_HEADS */

#if defined(__has_builtin)
#  if __has_builtin(__builtin_amdgcn_tensor_load_to_lds) && \
      __has_builtin(__builtin_amdgcn_s_wait_tensorcnt)
#    define USE_TDM 1
#  endif
#endif
#ifndef USE_TDM
#  define USE_TDM 0
#endif

__device__ __forceinline__ v8f wmma_bf16(v16bf a, v16bf b, v8f c) {
    return __builtin_amdgcn_wmma_f32_16x16x32_bf16(false, a, false, b,
                                                   (short)0, c, false, false);
}

// ---------------------------------------------------------------------------
// TDM: DMA a 2D bf16 tile [tile_d1 rows x tile_d0 elems] (row stride
// stride_elems in global) into LDS at lds_off, inserting pad_amount DWORDs
// of LDS padding after every pad_interval DWORDs (codes per D# group 1).
// Issued once per calling wave (EXEC-independent); gate with (wave==0).
// ---------------------------------------------------------------------------
__device__ __forceinline__ void tdm_load_bf16_2d(
    unsigned lds_off, const void* gptr,
    unsigned tile_d0, unsigned tile_d1,
    unsigned long long stride_elems,
    unsigned pad_interval_code, unsigned pad_amount_code)
{
#if USE_TDM
    unsigned long long ga = (unsigned long long)gptr;
    u32x4 g0;
    g0[0] = 1u;                                   // count=1, user descriptor
    g0[1] = lds_off;                              // LDS byte address
    g0[2] = (unsigned)ga;                         // global_addr[31:0]
    g0[3] = (unsigned)((ga >> 32) & 0x01FFFFFFu)  // global_addr[56:32]
            | (2u << 30);                         // type=2 ("image")
    i32x8 g1;
    g1[0] = (int)((1u << 16)                      // data_size = 2 bytes
                  | (1u << 20)                    // pad_enable
                  | (pad_interval_code << 22)
                  | (pad_amount_code << 25));
    g1[1] = (int)((tile_d0 & 0xFFFFu) << 16);     // tensor_dim0[15:0]
    g1[2] = (int)(((tile_d0 >> 16) & 0xFFFFu)     // tensor_dim0[31:16]
                  | ((tile_d1 & 0xFFFFu) << 16)); // tensor_dim1[15:0]
    g1[3] = (int)(((tile_d1 >> 16) & 0xFFFFu)     // tensor_dim1[31:16]
                  | (tile_d0 << 16));             // tile_dim0
    g1[4] = (int)(tile_d1 & 0xFFFFu);             // tile_dim1 (tile_dim2 = 0)
    g1[5] = (int)(unsigned)(stride_elems & 0xFFFFFFFFull);   // dim0_stride lo
    g1[6] = (int)(unsigned)((stride_elems >> 32) & 0xFFFFull); // dim0_stride hi
    g1[7] = 0;
    i32x4 z4 = {0, 0, 0, 0};
#if __clang_major__ >= 23
    i32x8 z8 = {0, 0, 0, 0, 0, 0, 0, 0};
    __builtin_amdgcn_tensor_load_to_lds(g0, g1, z4, z4, z8, 0);
#else
    __builtin_amdgcn_tensor_load_to_lds(g0, g1, z4, z4, 0);
#endif
#else
    (void)lds_off; (void)gptr; (void)tile_d0; (void)tile_d1;
    (void)stride_elems; (void)pad_interval_code; (void)pad_amount_code;
#endif
}

__device__ __forceinline__ void tdm_wait() {
#if USE_TDM
    __builtin_amdgcn_s_wait_tensorcnt(0);
#endif
}

// ---------------------------------------------------------------------------
// Weight conversion: 4x [1024x1024] f32 -> bf16, packed wq|wk|wv|wo
// ---------------------------------------------------------------------------
__global__ __launch_bounds__(256) void cvt_weights_kernel(
    const float* __restrict__ wq, const float* __restrict__ wk,
    const float* __restrict__ wv, const float* __restrict__ wo,
    bf16_t* __restrict__ dst)
{
    size_t i4 = ((size_t)blockIdx.x * 256 + threadIdx.x) * 4;
    const float* srcs[4] = {wq, wk, wv, wo};
    const float* s = srcs[i4 >> 20];
    float4 f = *(const float4*)(s + (i4 & ((1u << 20) - 1)));
    dst[i4 + 0] = (bf16_t)f.x;
    dst[i4 + 1] = (bf16_t)f.y;
    dst[i4 + 2] = (bf16_t)f.z;
    dst[i4 + 3] = (bf16_t)f.w;
}

// ---------------------------------------------------------------------------
// NT GEMM: C[M,N] = A[M,K] * W[N,K]^T + bias[N]
//   A_F32: A is f32 (converted to bf16 while staging), else A is bf16
//   OUT_F32: C is f32, else bf16
// Tile 128x128, K-step 32, double-buffered LDS. B (weight) tile staged by
// TDM DMA (pad 4 DWORDs per 16 DWORDs -> LDS stride 40 bf16); A staged by
// vector loads (needs f32->bf16 conversion).
// ---------------------------------------------------------------------------
template <bool A_F32, bool OUT_F32>
__global__ __launch_bounds__(256) void gemm_nt_kernel(
    const void* __restrict__ A_, const bf16_t* __restrict__ W,
    const float* __restrict__ bias, void* __restrict__ C_,
    int M, int N, int K)
{
    constexpr int BM = 128, BN = 128, BK = 32;
    constexpr int LDA = 40, LDB = 40;     // +8 bf16 pad: 16B aligned, bank-spread
    __shared__ bf16_t As[2][BM * LDA];
    __shared__ bf16_t Bs[2][BN * LDB];

    const int t    = threadIdx.x;
    const int lane = t & 31;
    const int w    = t >> 5;       // wave 0..7
    const int hl   = lane >> 4;    // half-wave: K-split selector per ISA layout
    const int l16  = lane & 15;
    const int wm   = w & 3;        // 4 row-groups of 32
    const int wn   = w >> 2;       // 2 col-groups of 64
    const int m0   = blockIdx.y * BM;
    const int n0   = blockIdx.x * BN;

    const int ar = t >> 1;         // staging row 0..127
    const int ac = (t & 1) << 4;   // staging k-offset {0,16}

    auto stage_A = [&](int kt, int buf) {
        if constexpr (A_F32) {
            const float* ap = (const float*)A_ + (size_t)(m0 + ar) * K + kt + ac;
            __builtin_prefetch(ap + 2 * BK, 0, 0);
            float4 f0 = ((const float4*)ap)[0];
            float4 f1 = ((const float4*)ap)[1];
            float4 f2 = ((const float4*)ap)[2];
            float4 f3 = ((const float4*)ap)[3];
            bf16_t* d = &As[buf][ar * LDA + ac];
            d[0]  = (bf16_t)f0.x; d[1]  = (bf16_t)f0.y; d[2]  = (bf16_t)f0.z; d[3]  = (bf16_t)f0.w;
            d[4]  = (bf16_t)f1.x; d[5]  = (bf16_t)f1.y; d[6]  = (bf16_t)f1.z; d[7]  = (bf16_t)f1.w;
            d[8]  = (bf16_t)f2.x; d[9]  = (bf16_t)f2.y; d[10] = (bf16_t)f2.z; d[11] = (bf16_t)f2.w;
            d[12] = (bf16_t)f3.x; d[13] = (bf16_t)f3.y; d[14] = (bf16_t)f3.z; d[15] = (bf16_t)f3.w;
        } else {
            const uint4* ap = (const uint4*)((const bf16_t*)A_ +
                               (size_t)(m0 + ar) * K + kt + ac);
            uint4* d = (uint4*)&As[buf][ar * LDA + ac];
            d[0] = ap[0]; d[1] = ap[1];
        }
    };

    auto stage_B = [&](int kt, int buf) {
#if USE_TDM
        if (w == 0) {
            tdm_load_bf16_2d((unsigned)(unsigned long long)&Bs[buf][0],
                             W + (size_t)n0 * K + kt,
                             /*tile_d0=*/BK, /*tile_d1=*/BN,
                             /*stride=*/(unsigned long long)K,
                             /*pad_interval 16DW=*/3, /*pad 4DW=*/3);
        }
#else
        const uint4* wp = (const uint4*)(W + (size_t)(n0 + ar) * K + kt + ac);
        uint4* db = (uint4*)&Bs[buf][ar * LDB + ac];
        db[0] = wp[0]; db[1] = wp[1];
#endif
    };

    v8f acc[2][4] = {};

    stage_A(0, 0);
    stage_B(0, 0);
    tdm_wait();
    __syncthreads();

    const int nk = K / BK;
    for (int s = 0; s < nk; ++s) {
        const int buf = s & 1;
        if (s + 1 < nk) {
            stage_A((s + 1) * BK, buf ^ 1);
            stage_B((s + 1) * BK, buf ^ 1);
        }

        union { v16bf v; uint4 q[2]; } au[2];
#pragma unroll
        for (int i = 0; i < 2; ++i) {
            // A fragment 16x32 bf16: lane<16 holds K 0..15,
            // lane>=16 holds K 16..31
            const bf16_t* ap = &As[buf][(wm * 32 + i * 16 + l16) * LDA + hl * 16];
            au[i].q[0] = *(const uint4*)ap;
            au[i].q[1] = *(const uint4*)(ap + 8);
        }
#pragma unroll
        for (int j = 0; j < 4; ++j) {
            // B fragment 32x16 bf16: lane<16 -> K 0..15, lane>=16 -> K 16..31
            union { v16bf v; uint4 q[2]; } bu;
            const bf16_t* bp = &Bs[buf][(wn * 64 + j * 16 + l16) * LDB + hl * 16];
            bu.q[0] = *(const uint4*)bp;
            bu.q[1] = *(const uint4*)(bp + 8);
#pragma unroll
            for (int i = 0; i < 2; ++i)
                acc[i][j] = wmma_bf16(au[i].v, bu.v, acc[i][j]);
        }
        tdm_wait();
        __syncthreads();
    }

    // Epilogue: C/D layout -> lane holds col l16, rows (hl*8 + r)
#pragma unroll
    for (int i = 0; i < 2; ++i) {
#pragma unroll
        for (int j = 0; j < 4; ++j) {
            const int gm = m0 + wm * 32 + i * 16 + hl * 8;
            const int gn = n0 + wn * 64 + j * 16 + l16;
            const float bv = bias[gn];
#pragma unroll
            for (int r = 0; r < 8; ++r) {
                const float v = acc[i][j][r] + bv;
                if constexpr (OUT_F32)
                    ((float*)C_)[(size_t)(gm + r) * N + gn] = v;
                else
                    ((bf16_t*)C_)[(size_t)(gm + r) * N + gn] = (bf16_t)v;
            }
        }
    }
}

// ---------------------------------------------------------------------------
// Attention: one block per (b, h, 128-query-row slab).
// Head view quirk: q[b,h,i,:] = Qproj[b, i*16+h, :]  (rows strided by 16).
// Phase 1: scores 128x256 over K=1024 -> registers; Q/K tiles TDM-staged
//          (row stride 16*1024 elems), double-buffered.
// Phase 2: register softmax (shfl_xor within 16-lane halves) ->
//          attn f32 to d_out, bf16 to LDS
// Phase 3: ctx[128x1024] = attn * V
// ---------------------------------------------------------------------------
__global__ __launch_bounds__(256) void attn_softmax_ctx_kernel(
    const bf16_t* __restrict__ qb, const bf16_t* __restrict__ kb,
    const bf16_t* __restrict__ vb, float* __restrict__ attn_out,
    bf16_t* __restrict__ ctx)
{
    constexpr int LQ = 40, LK = 40, LATT = 264, LV = 72;
    constexpr int QBUF = 128 * LQ;               // 5120 elems per Q buffer
    constexpr int KBUF = 256 * LK;               // 10240 elems per K buffer
    __shared__ bf16_t smem[36096];               // 72 KB, phases overlaid
    // phase-1 overlay: qs[buf] = smem + buf*QBUF ; ks[buf] = smem + 2*QBUF + buf*KBUF
    // phase-3 overlay: attnbf = smem ; Vs = smem + 128*LATT
    bf16_t* attnbf = smem;                       // 128*264
    bf16_t* Vs     = smem + 128 * LATT;          // 32*72

    const int t    = threadIdx.x;
    const int lane = t & 31;
    const int w    = t >> 5;
    const int hl   = lane >> 4;
    const int l16  = lane & 15;
    const int rb   = blockIdx.x;     // 0..1  (query row slab)
    const int h    = blockIdx.y;     // head
    const int b    = blockIdx.z;     // batch

    auto qs_at = [&](int buf) -> bf16_t* { return smem + buf * QBUF; };
    auto ks_at = [&](int buf) -> bf16_t* { return smem + 2 * QBUF + buf * KBUF; };

    auto qrow = [&](int lr) {
        return (size_t)b * SEQ + (size_t)(rb * 128 + lr) * NUM_HEADS + h;
    };
    auto krow = [&](int kr) {
        return (size_t)b * SEQ + (size_t)kr * NUM_HEADS + h;
    };

    auto stage_qk = [&](int kt, int buf) {
#if USE_TDM
        if (w == 0) {
            tdm_load_bf16_2d((unsigned)(unsigned long long)qs_at(buf),
                             qb + qrow(0) * KEY_SIZE + kt,
                             32, 128, (unsigned long long)NUM_HEADS * KEY_SIZE,
                             3, 3);
            tdm_load_bf16_2d((unsigned)(unsigned long long)ks_at(buf),
                             kb + krow(0) * KEY_SIZE + kt,
                             32, 256, (unsigned long long)NUM_HEADS * KEY_SIZE,
                             3, 3);
        }
#else
        {   // Q slab rows: each thread 32B
            const int sr = t >> 1, sc = (t & 1) << 4;
            const uint4* src = (const uint4*)(qb + qrow(sr) * KEY_SIZE + kt + sc);
            uint4* d = (uint4*)&qs_at(buf)[sr * LQ + sc];
            d[0] = src[0]; d[1] = src[1];
        }
        {   // all 256 key rows, 32 k each
            const uint4* src = (const uint4*)(kb + krow(t) * KEY_SIZE + kt);
            uint4* d = (uint4*)&ks_at(buf)[t * LK];
            d[0] = src[0]; d[1] = src[1]; d[2] = src[2]; d[3] = src[3];
        }
#endif
    };

    // ---------------- Phase 1: scores ----------------
    v8f acc[16] = {};
    stage_qk(0, 0);
    tdm_wait();
    __syncthreads();

    for (int kt = 0; kt < KEY_SIZE; kt += 32) {
        const int buf = (kt >> 5) & 1;
        if (kt + 32 < KEY_SIZE) stage_qk(kt + 32, buf ^ 1);

        union { v16bf v; uint4 q[2]; } au;
        const bf16_t* ap = &qs_at(buf)[(w * 16 + l16) * LQ + hl * 16];
        au.q[0] = *(const uint4*)ap;
        au.q[1] = *(const uint4*)(ap + 8);
#pragma unroll
        for (int j = 0; j < 16; ++j) {
            union { v16bf v; uint4 q[2]; } bu;
            const bf16_t* bp = &ks_at(buf)[(j * 16 + l16) * LK + hl * 16];
            bu.q[0] = *(const uint4*)bp;
            bu.q[1] = *(const uint4*)(bp + 8);
            acc[j] = wmma_bf16(au.v, bu.v, acc[j]);
        }
        tdm_wait();
        __syncthreads();
    }

    // ---------------- Phase 2: register softmax ----------------
    const float scale = 0.125f;                 // 1/sqrt(HEAD_DIM=64)
    float rinv[8];
#pragma unroll
    for (int r = 0; r < 8; ++r) {
        float m = -3.0e38f;
#pragma unroll
        for (int j = 0; j < 16; ++j) {
            acc[j][r] *= scale;
            m = fmaxf(m, acc[j][r]);
        }
#pragma unroll
        for (int off = 8; off; off >>= 1)       // reduce within 16-lane half
            m = fmaxf(m, __shfl_xor(m, off, 32));
        float s = 0.f;
#pragma unroll
        for (int j = 0; j < 16; ++j) {
            const float e = __expf(acc[j][r] - m);
            acc[j][r] = e;
            s += e;
        }
#pragma unroll
        for (int off = 8; off; off >>= 1)
            s += __shfl_xor(s, off, 32);
        rinv[r] = 1.0f / s;
    }

    const size_t attn_base = (((size_t)b * NUM_HEADS + h) * SQ + rb * 128) * SQ;
#pragma unroll
    for (int j = 0; j < 16; ++j) {
        const int col = j * 16 + l16;
#pragma unroll
        for (int r = 0; r < 8; ++r) {
            const int row = w * 16 + hl * 8 + r;
            const float p = acc[j][r] * rinv[r];
            attn_out[attn_base + (size_t)row * SQ + col] = p;
            attnbf[row * LATT + col] = (bf16_t)p;
        }
    }
    __syncthreads();

    // ---------------- Phase 3: ctx = attn * V ----------------
    const int vk = t >> 3;
    const int vc = (t & 7) << 3;
    for (int nc = 0; nc < KEY_SIZE; nc += 64) {
        v8f cacc[4] = {};
        for (int k0 = 0; k0 < SQ; k0 += 32) {
            // stage V[k0..k0+32, nc..nc+64] as [k][n]
            const uint4* src =
                (const uint4*)(vb + krow(k0 + vk) * KEY_SIZE + nc + vc);
            *(uint4*)&Vs[vk * LV + vc] = src[0];
            __syncthreads();

            union { v16bf v; uint4 q[2]; } au;
            const bf16_t* ap = &attnbf[(w * 16 + l16) * LATT + k0 + hl * 16];
            au.q[0] = *(const uint4*)ap;
            au.q[1] = *(const uint4*)(ap + 8);
#pragma unroll
            for (int j = 0; j < 4; ++j) {
                v16bf bv;
#pragma unroll
                for (int e = 0; e < 16; ++e)
                    bv[e] = Vs[(hl * 16 + e) * LV + j * 16 + l16];
                cacc[j] = wmma_bf16(au.v, bv, cacc[j]);
            }
            __syncthreads();
        }
#pragma unroll
        for (int j = 0; j < 4; ++j) {
#pragma unroll
            for (int r = 0; r < 8; ++r) {
                const int lr = w * 16 + hl * 8 + r;
                ctx[qrow(lr) * KEY_SIZE + nc + j * 16 + l16] = (bf16_t)cacc[j][r];
            }
        }
    }
}

// ---------------------------------------------------------------------------
extern "C" void kernel_launch(void* const* d_in, const int* in_sizes, int n_in,
                              void* d_out, int out_size, void* d_ws, size_t ws_size,
                              hipStream_t stream) {
    const float* query = (const float*)d_in[0];
    const float* key_  = (const float*)d_in[1];
    const float* value = (const float*)d_in[2];
    const float* Wq = (const float*)d_in[3];
    const float* bq = (const float*)d_in[4];
    const float* Wk = (const float*)d_in[5];
    const float* bk = (const float*)d_in[6];
    const float* Wv = (const float*)d_in[7];
    const float* bv = (const float*)d_in[8];
    const float* Wo = (const float*)d_in[9];
    const float* bo = (const float*)d_in[10];

    // Workspace layout (bf16 elements)
    const size_t WSZ  = (size_t)KEY_SIZE * D_MODEL;   // 1M per weight
    const size_t ACTS = (size_t)MROWS * KEY_SIZE;     // 33.5M per activation
    bf16_t* wqb = (bf16_t*)d_ws;
    bf16_t* wkb = wqb + WSZ;
    bf16_t* wvb = wkb + WSZ;
    bf16_t* wob = wvb + WSZ;
    bf16_t* qb  = wob + WSZ;
    bf16_t* kb  = qb + ACTS;
    bf16_t* vb  = kb + ACTS;
    bf16_t* ctx = vb + ACTS;

    float* out_proj = (float*)d_out;                       // [B,S,1024] f32
    float* attn_out = out_proj + (size_t)MROWS * KEY_SIZE; // [B,H,256,256] f32

    // 1) weights -> bf16
    cvt_weights_kernel<<<4096, 256, 0, stream>>>(Wq, Wk, Wv, Wo, wqb);

    // 2) q/k/v projections (f32 act -> bf16 out)
    dim3 gg(KEY_SIZE / 128, MROWS / 128);   // (8, 256)
    gemm_nt_kernel<true, false><<<gg, 256, 0, stream>>>(
        query, wqb, bq, qb, MROWS, KEY_SIZE, D_MODEL);
    gemm_nt_kernel<true, false><<<gg, 256, 0, stream>>>(
        key_, wkb, bk, kb, MROWS, KEY_SIZE, D_MODEL);
    gemm_nt_kernel<true, false><<<gg, 256, 0, stream>>>(
        value, wvb, bv, vb, MROWS, KEY_SIZE, D_MODEL);

    // 3) attention (scores + softmax + context), writes attn f32 directly
    attn_softmax_ctx_kernel<<<dim3(2, NUM_HEADS, BATCH), 256, 0, stream>>>(
        qb, kb, vb, attn_out, ctx);

    // 4) output projection (bf16 act -> f32 out)
    gemm_nt_kernel<false, true><<<gg, 256, 0, stream>>>(
        ctx, wob, bo, out_proj, MROWS, D_MODEL, KEY_SIZE);
}
